// GNN_8057358647777
// MI455X (gfx1250) — matrix operand order repaired and checked by reading, not compiled, not run on hardware
//
#include <hip/hip_runtime.h>
#include <hip/hip_bf16.h>

typedef __attribute__((ext_vector_type(16))) __bf16 v16bf;
typedef __attribute__((ext_vector_type(8)))  float  v8f;

union U16 { v16bf v; uint4 u[2]; };
union U4  { __bf16 b[4]; uint2 u; };

__device__ __forceinline__ float4 ld4(const float* p) { return *(const float4*)p; }

// ---------------------------------------------------------------------------
// GEMM: out[M,Nout] = A @ W^T + bias, A is [M,K] bf16, W is [Nout,K] bf16.
// If colsum != null: accumulates per-column sum / sum-of-squares of out
// (training-mode BatchNorm statistics) via wave-reduced atomics.
// Block = 256 threads = 8 waves; wave computes 16x64 via 4 WMMA accumulators;
// block tile = 128 rows x 64 cols. The 64xK bf16 weight panel is staged in
// LDS once per block (<= 64 KB for K=512) and shared by all 8 waves.
// Out-of-range rows load a clamped row (results masked at store/stat time),
// keeping the K-loop free of divergence and exec-mask churn.
// ---------------------------------------------------------------------------
__global__ __launch_bounds__(256)
void gin_gemm_wmma(const __bf16* __restrict__ A,
                   const __bf16* __restrict__ W,
                   const float* __restrict__ bias,
                   float* __restrict__ out,
                   float* __restrict__ colsum,
                   float* __restrict__ colsumsq,
                   int M, int K, int Nout)
{
    __shared__ __bf16 Bs[64 * 512];

    const int tid     = threadIdx.x;
    const int lane    = tid & 31;
    const int wave    = tid >> 5;
    const int nColBlk = Nout >> 6;                  // 64-wide column blocks
    const int bm      = blockIdx.x / nColBlk;
    const int bn      = blockIdx.x - bm * nColBlk;
    const int rowBase = bm * 128 + wave * 16;
    const int colBase = bn * 64;

    // ---- cooperative stage of the 64 x K B-panel into LDS (b128 chunks) ----
    const int kc = K >> 3;                          // 8-elem (16B) chunks per col
    for (int ch = tid; ch < 64 * kc; ch += 256) {
        const int c  = ch / kc;
        const int k8 = (ch - c * kc) << 3;
        *(uint4*)(Bs + c * K + k8) =
            *(const uint4*)(W + (size_t)(colBase + c) * K + k8);
    }
    __syncthreads();

    const int laneRow = lane & 15;                  // M (for A/C) or N (for B)
    const int hi      = lane >> 4;                  // which K-half this lane holds
    const int row     = rowBase + laneRow;
    const int rowCl   = (row < M) ? row : (M - 1);  // clamped: loads always legal
    const __bf16* ap  = A + (size_t)rowCl * K;

    v8f acc[4];
#pragma unroll
    for (int t = 0; t < 4; ++t)
        acc[t] = (v8f){0.f, 0.f, 0.f, 0.f, 0.f, 0.f, 0.f, 0.f};

    for (int k0 = 0; k0 < K; k0 += 32) {
        // ---- A fragment: 16x32 bf16 (ISA 7.12.2 16-bit A layout) ----
        // lane<16 : elems 0..7 = K k0+0..7,  elems 8..15 = K k0+16..23
        // lane>=16: elems 0..7 = K k0+8..15, elems 8..15 = K k0+24..31
        const int kbA = k0 + hi * 8;
        U16 au;
        au.u[0] = *(const uint4*)(ap + kbA);        // 8 bf16, one b128
        au.u[1] = *(const uint4*)(ap + kbA + 16);

        // ---- B fragments from LDS: lane<16 holds K k0+0..7,16..23, lane>=16 K+8..15,24..31
        const int kbB = k0 + hi * 8;
#pragma unroll
        for (int t = 0; t < 4; ++t) {
            const __bf16* bp = Bs + (size_t)(16 * t + laneRow) * K + kbB;
            U16 bu;
            bu.u[0] = *(const uint4*)bp;                // ds_load_b128
            bu.u[1] = *(const uint4*)(bp + 16);
            acc[t] = __builtin_amdgcn_wmma_f32_16x16x32_bf16(
                false, au.v, false, bu.v, (short)0, acc[t], false, false);
        }
    }

    // ---- epilogue: bias, store, BN column statistics ----
    // C layout: VGPR j, lanes 0-15 -> M=j, lanes 16-31 -> M=j+8; N = lane%16
    const int rowOff = hi * 8;
#pragma unroll
    for (int t = 0; t < 4; ++t) {
        const int col = colBase + 16 * t + laneRow;
        const float bv = (bias != nullptr) ? bias[col] : 0.f;
        float csum = 0.f, csq = 0.f;
#pragma unroll
        for (int j = 0; j < 8; ++j) {
            const int r = rowBase + rowOff + j;
            if (r < M) {
                const float v = acc[t][j] + bv;
                out[(size_t)r * Nout + col] = v;
                csum += v;
                csq  += v * v;
            }
        }
        if (colsum != nullptr) {
            // lanes L and L^16 hold the same column -> combine, one atomic each
            csum += __shfl_xor(csum, 16);
            csq  += __shfl_xor(csq, 16);
            if (lane < 16) {
                atomicAdd(&colsum[col],   csum);
                atomicAdd(&colsumsq[col], csq);
            }
        }
    }
}

// ---------------------------------------------------------------------------
// Edge scatter: agg[dst] += relu(x[src]); 64 lanes per edge, float4 per lane.
// ---------------------------------------------------------------------------
__global__ __launch_bounds__(256)
void gin_scatter(const float* __restrict__ x, const int* __restrict__ ei,
                 float* __restrict__ agg, int E)
{
    const int e = blockIdx.x * 4 + (threadIdx.x >> 6);
    if (e >= E) return;
    const int c   = (threadIdx.x & 63) * 4;
    const int src = ei[e];
    const int dst = ei[E + e];
    const float4 v = ld4(x + (size_t)src * 256 + c);
    float* d = agg + (size_t)dst * 256 + c;
    atomicAdd(d + 0, fmaxf(v.x, 0.f));
    atomicAdd(d + 1, fmaxf(v.y, 0.f));
    atomicAdd(d + 2, fmaxf(v.z, 0.f));
    atomicAdd(d + 3, fmaxf(v.w, 0.f));
}

// ---------------------------------------------------------------------------
// GIN combine: out_bf16 = (1+eps)*x + agg, packed 4-wide.
// ---------------------------------------------------------------------------
__global__ __launch_bounds__(256)
void gin_combine(const float* __restrict__ x, const float* __restrict__ agg,
                 const float* __restrict__ epsP, __bf16* __restrict__ out,
                 size_t n4)
{
    const float ep1 = 1.0f + epsP[0];
    size_t i = (size_t)blockIdx.x * blockDim.x + threadIdx.x;
    const size_t stride = (size_t)gridDim.x * blockDim.x;
    for (; i < n4; i += stride) {
        const float4 a = ld4(x + 4 * i);
        const float4 g = ld4(agg + 4 * i);
        U4 o;
        o.b[0] = (__bf16)fmaf(ep1, a.x, g.x);
        o.b[1] = (__bf16)fmaf(ep1, a.y, g.y);
        o.b[2] = (__bf16)fmaf(ep1, a.z, g.z);
        o.b[3] = (__bf16)fmaf(ep1, a.w, g.w);
        *(uint2*)(out + 4 * i) = o.u;
    }
}

// ---------------------------------------------------------------------------
// BN apply: out = (h - mu) * rsqrt(var+1e-5) * gamma + beta  (+ optional ReLU)
// using precomputed column sum / sumsq. C must be a power of two.
// OT = float (layer output) or __bf16 (feeds next GEMM directly).
// ---------------------------------------------------------------------------
template <typename OT>
__global__ __launch_bounds__(256)
void gin_bn_apply(const float* __restrict__ h, const float* __restrict__ sum,
                  const float* __restrict__ sumsq, const float* __restrict__ gamma,
                  const float* __restrict__ beta, OT* __restrict__ out,
                  size_t total, int Cmask, float invM, int doRelu)
{
    size_t i = (size_t)blockIdx.x * blockDim.x + threadIdx.x;
    const size_t stride = (size_t)gridDim.x * blockDim.x;
    for (; i < total; i += stride) {
        const int   c   = (int)(i & (size_t)Cmask);
        const float mu  = sum[c] * invM;
        const float var = fmaxf(sumsq[c] * invM - mu * mu, 0.f);
        const float sc  = __frsqrt_rn(var + 1e-5f) * gamma[c];
        float v = (h[i] - mu) * sc + beta[c];
        if (doRelu) v = fmaxf(v, 0.f);
        out[i] = (OT)v;
    }
}

__global__ __launch_bounds__(256)
void cvt_f32_bf16(const float* __restrict__ in, __bf16* __restrict__ out, size_t n)
{
    size_t i = (size_t)blockIdx.x * blockDim.x + threadIdx.x;
    const size_t stride = (size_t)gridDim.x * blockDim.x;
    for (; i < n; i += stride) out[i] = (__bf16)in[i];
}

__global__ __launch_bounds__(256)
void gin_zero(float* __restrict__ p, size_t n)
{
    size_t i = (size_t)blockIdx.x * blockDim.x + threadIdx.x;
    const size_t stride = (size_t)gridDim.x * blockDim.x;
    for (; i < n; i += stride) p[i] = 0.f;
}

extern "C" void kernel_launch(void* const* d_in, const int* in_sizes, int n_in,
                              void* d_out, int out_size, void* d_ws, size_t ws_size,
                              hipStream_t stream)
{
    const float* x    = (const float*)d_in[0];
    const int*   ei   = (const int*)  d_in[1];
    const float* fc_w = (const float*)d_in[2];
    const float* fc_b = (const float*)d_in[3];
    const float* eps  = (const float*)d_in[4];
    const float* W1   = (const float*)d_in[5];
    const float* b1   = (const float*)d_in[6];
    const float* g1   = (const float*)d_in[7];
    const float* be1  = (const float*)d_in[8];
    const float* W2   = (const float*)d_in[9];
    const float* b2   = (const float*)d_in[10];
    const float* gbn  = (const float*)d_in[11];
    const float* bbn  = (const float*)d_in[12];

    const int N  = in_sizes[0] / 32;   // 50000 nodes
    const int E  = in_sizes[1] / 2;    // 800000 edges
    const int D  = 256;
    const int D2 = 512;
    const float invM = 1.0f / (float)N;

    // ---- workspace layout ----
    float* xcur  = (float*)d_ws;                       // [N,256] f32
    float* agg   = xcur + (size_t)N * D;               // [N,256] f32 (alias h2)
    float* h1    = agg  + (size_t)N * D;               // [N,512] f32
    float* s1    = h1   + (size_t)N * D2;              // [512]
    float* ss1   = s1  + D2;                           // [512]
    float* s2    = ss1 + D2;                           // [256]
    float* ss2   = s2  + D;                            // [256]
    float* h2    = agg;                                // alias
    __bf16* actb = (__bf16*)(ss2 + D);                 // [N,512] bf16 (time-shared)
    __bf16* xb   = actb + (size_t)N * D2;              // [N,32]  bf16
    __bf16* fcwb = xb   + (size_t)N * 32;              // [256,32]
    __bf16* W1b  = fcwb + (size_t)D * 32;              // [3,512,256]
    __bf16* W2b  = W1b  + (size_t)3 * D2 * D;          // [3,256,512]

    const dim3 blk(256);
    auto gemmGrid = [](int M, int Nout) { return dim3(((M + 127) / 128) * (Nout / 64)); };

    // ---- one-time bf16 conversions (inputs never mutated) ----
    cvt_f32_bf16<<<1024, 256, 0, stream>>>(x,    xb,   (size_t)N * 32);
    cvt_f32_bf16<<<32,   256, 0, stream>>>(fc_w, fcwb, (size_t)D * 32);
    cvt_f32_bf16<<<512,  256, 0, stream>>>(W1,   W1b,  (size_t)3 * D2 * D);
    cvt_f32_bf16<<<512,  256, 0, stream>>>(W2,   W2b,  (size_t)3 * D * D2);

    // fc_in: xcur = x @ fc_w^T + fc_b  (K=32)
    gin_gemm_wmma<<<gemmGrid(N, D), blk, 0, stream>>>(
        xb, fcwb, fc_b, xcur, nullptr, nullptr, N, 32, D);

    for (int l = 0; l < 3; ++l) {
        gin_zero<<<1024, 256, 0, stream>>>(agg, (size_t)N * D);
        gin_zero<<<2, 256, 0, stream>>>(s1, (size_t)(2 * D2 + 2 * D));

        gin_scatter<<<(E + 3) / 4, 256, 0, stream>>>(xcur, ei, agg, E);

        // actb[:,256] = bf16((1+eps)*x + agg)
        gin_combine<<<2048, 256, 0, stream>>>(xcur, agg, eps + l, actb,
                                              (size_t)N * D / 4);

        // h1 = actb @ W1^T + b1   (+ BN1 stats)
        gin_gemm_wmma<<<gemmGrid(N, D2), blk, 0, stream>>>(
            actb, W1b + (size_t)l * D2 * D, b1 + (size_t)l * D2,
            h1, s1, ss1, N, D, D2);

        // actb[:,512] = bf16(relu(BN1(h1)))
        gin_bn_apply<__bf16><<<4096, 256, 0, stream>>>(
            h1, s1, ss1, g1 + (size_t)l * D2, be1 + (size_t)l * D2, actb,
            (size_t)N * D2, D2 - 1, invM, 1);

        // h2 = actb @ W2^T + b2   (+ BN2 stats)
        gin_gemm_wmma<<<gemmGrid(N, D), blk, 0, stream>>>(
            actb, W2b + (size_t)l * D * D2, b2 + (size_t)l * D,
            h2, s2, ss2, N, D2, D);

        // x = BN2(h2) (+ relu except last layer); last layer -> d_out
        float* dst = (l == 2) ? (float*)d_out : xcur;
        gin_bn_apply<float><<<4096, 256, 0, stream>>>(
            h2, s2, ss2, gbn + (size_t)l * D, bbn + (size_t)l * D, dst,
            (size_t)N * D, D - 1, invM, (l < 2) ? 1 : 0);
    }
}
